// MaskedTransformerEmbeddingCosine_25031069401603
// MI455X (gfx1250) — compile-verified, matrix operand order checked by reading, not run on hardware
//
#include <hip/hip_runtime.h>
#include <hip/hip_bf16.h>
#include <math.h>
#include <stdint.h>

// ---------------- problem constants (from reference) ----------------
#define BB 8
#define LL 1024
#define DD 640
#define HH 10
#define FF 1280
#define NLAYER 6
#define DHD 64
#define WBAND 64
#define MTOK (BB * LL)        // 8192 token rows
#define KT 9                  // key tiles per query tile in the band
#define KPAD 160              // 5 pairs * 32 keys (padded)
#define KCHUNK 128            // K staged per LDS panel in GEMM

// ---------------- WMMA / TDM types ----------------
typedef __attribute__((ext_vector_type(16))) __bf16 v16bf;
typedef __attribute__((ext_vector_type(8)))  float  v8f;
typedef __attribute__((ext_vector_type(4)))  unsigned int v4u;
typedef __attribute__((ext_vector_type(8)))  int v8i;
typedef __attribute__((ext_vector_type(4)))  int v4i;

#if defined(__HIP_DEVICE_COMPILE__) && __has_builtin(__builtin_amdgcn_tensor_load_to_lds)
#define HAVE_TDM 1
#else
#define HAVE_TDM 0
#endif

struct Frag32 { uint32_t u[8]; };   // 32 bytes == v16bf

static __device__ inline Frag32 mkfrag(uint4 lo, uint4 hi) {
  Frag32 f;
  f.u[0] = lo.x; f.u[1] = lo.y; f.u[2] = lo.z; f.u[3] = lo.w;
  f.u[4] = hi.x; f.u[5] = hi.y; f.u[6] = hi.z; f.u[7] = hi.w;
  return f;
}

static __device__ inline v8f wmma_bf16(const Frag32& a, const Frag32& b, v8f c) {
  return __builtin_amdgcn_wmma_f32_16x16x32_bf16(
      false, __builtin_bit_cast(v16bf, a),
      false, __builtin_bit_cast(v16bf, b),
      (short)0, c, false, false);
}

static __device__ inline unsigned short f32_to_bf16u(float x) {
  uint32_t b = __builtin_bit_cast(uint32_t, x);
  uint32_t r = b + 0x7FFFu + ((b >> 16) & 1u);   // round-to-nearest-even
  return (unsigned short)(r >> 16);
}

// ---------------- fp32 -> bf16 convert ----------------
__global__ void k_cvt_bf16(const float* __restrict__ src,
                           unsigned short* __restrict__ dst, size_t n) {
  size_t i = (size_t)blockIdx.x * blockDim.x + threadIdx.x;
  size_t stride = (size_t)gridDim.x * blockDim.x;
  for (; i < n; i += stride) dst[i] = f32_to_bf16u(src[i]);
}

// ---------------- tiled bf16 WMMA GEMM: out = act(A[MxK] * W[NxK]^T + bias) ----
// Block: 256 thr = 8 waves as 2(M) x 4(N); each wave owns two 16x16 acc tiles
// (N and N+64). A panel 32xKCHUNK staged in LDS (TDM or cooperative b128).
// 8 WMMAs per wave per barrier pair.
#define GEMM_RELU 1
#define GEMM_BF16OUT 2

__global__ __launch_bounds__(256) void k_gemm_bf16(
    const unsigned short* __restrict__ A,   // M x K row-major bf16
    const unsigned short* __restrict__ W,   // N x K row-major bf16
    const float* __restrict__ bias,         // N (may be null)
    float* __restrict__ outF,               // M x N fp32 (if !BF16OUT)
    unsigned short* __restrict__ outB,      // M x N bf16 (if BF16OUT)
    int M, int N, int K, int flags) {
  __shared__ __align__(16) unsigned short Atile[32][KCHUNK];  // 8 KB

  const int tid  = threadIdx.x;
  const int wave = tid >> 5;
  const int lane = tid & 31;
  const int lrow = lane & 15;
  const int khalf = (lane >= 16) ? 8 : 0;   // CDNA5 bf16 operand K-interleave
  const int mr = wave >> 2;                 // 0..1  (M sub-row)
  const int wc = wave & 3;                  // 0..3  (N sub-col)

  const int mbase = blockIdx.x * 32;
  const int nb0 = blockIdx.y * 128 + wc * 16;
  const int nb1 = nb0 + 64;
  const unsigned short* wrow0 = W + (size_t)(nb0 + lrow) * K;
  const unsigned short* wrow1 = W + (size_t)(nb1 + lrow) * K;

  // cooperative-staging coordinates (fallback path)
  const int srow = tid >> 3;                // 0..31
  const int scol = (tid & 7) * 16;          // 0..112

  v8f acc0 = {}, acc1 = {};
  for (int k0 = 0; k0 < K; k0 += KCHUNK) {
#if HAVE_TDM
    if (wave == 0) {
      // Tensor Data Mover: DMA a 32 x 128 bf16 tile (row stride K) into LDS.
      const uint64_t ga = (uint64_t)(uintptr_t)(A + (size_t)mbase * K + k0);
      const uint32_t lds = (uint32_t)(uintptr_t)&Atile[0][0];
      v4u g0;
      g0[0] = 1u;                                   // count=1 valid descriptor
      g0[1] = lds;                                  // lds_addr
      g0[2] = (uint32_t)ga;                         // global_addr[31:0]
      g0[3] = (uint32_t)((ga >> 32) & 0x1FFFFFFu) | (2u << 30);  // [56:32]|type=2
      v8i g1;
      g1[0] = 1 << 16;                              // data_size=1 (2 bytes/elem)
      g1[1] = (K & 0xFFFF) << 16;                   // tensor_dim0[15:0]
      g1[2] = ((K >> 16) & 0xFFFF) | (32 << 16);    // tensor_dim0[31:16] | tensor_dim1
      g1[3] = KCHUNK << 16;                         // tile_dim0 = 128
      g1[4] = 32;                                   // tile_dim1 = 32 (tile_dim2=0)
      g1[5] = K;                                    // tensor_dim0_stride[31:0]
      g1[6] = 0;                                    // stride hi / dim1_stride lo
      g1[7] = 0;
      v4i z4 = {0, 0, 0, 0};
#if defined(__clang_major__) && (__clang_major__ >= 23)
      v8i z8 = {0, 0, 0, 0, 0, 0, 0, 0};
      __builtin_amdgcn_tensor_load_to_lds(g0, g1, z4, z4, z8, 0);
#else
      __builtin_amdgcn_tensor_load_to_lds(g0, g1, z4, z4, 0);
#endif
      __builtin_amdgcn_s_wait_tensorcnt(0);
    }
#else
    {
      const unsigned short* src = A + (size_t)(mbase + srow) * K + k0 + scol;
      *(uint4*)&Atile[srow][scol]     = *(const uint4*)&src[0];
      *(uint4*)&Atile[srow][scol + 8] = *(const uint4*)&src[8];
    }
#endif
    __syncthreads();

    if (k0 + KCHUNK < K) {
      __builtin_prefetch(&wrow0[k0 + KCHUNK], 0, 1);
      __builtin_prefetch(&wrow1[k0 + KCHUNK], 0, 1);
    }
#pragma unroll
    for (int ks = 0; ks < KCHUNK; ks += 32) {
      const Frag32 fa = mkfrag(*(const uint4*)&Atile[mr * 16 + lrow][ks + khalf],
                               *(const uint4*)&Atile[mr * 16 + lrow][ks + khalf + 16]);
      const Frag32 fb0 = mkfrag(*(const uint4*)&wrow0[k0 + ks + khalf],
                                *(const uint4*)&wrow0[k0 + ks + khalf + 16]);
      const Frag32 fb1 = mkfrag(*(const uint4*)&wrow1[k0 + ks + khalf],
                                *(const uint4*)&wrow1[k0 + ks + khalf + 16]);
      acc0 = wmma_bf16(fa, fb0, acc0);
      acc1 = wmma_bf16(fa, fb1, acc1);
    }
    __syncthreads();
  }

  // epilogue: C layout -> lanes<16: M=r, lanes>=16: M=8+r, N=lane&15
  const int mrb = mbase + mr * 16 + ((lane >= 16) ? 8 : 0);
#pragma unroll
  for (int t = 0; t < 2; ++t) {
    const v8f acc = t ? acc1 : acc0;
    const int nn = (t ? nb1 : nb0) + lrow;
    const float bv = bias ? bias[nn] : 0.0f;
#pragma unroll
    for (int r = 0; r < 8; ++r) {
      float v = acc[r] + bv;
      if (flags & GEMM_RELU) v = fmaxf(v, 0.0f);
      size_t off = (size_t)(mrb + r) * N + nn;
      if (flags & GEMM_BF16OUT) outB[off] = f32_to_bf16u(v);
      else                      outF[off] = v;
    }
  }
}

// ---------------- banded attention, one wave per (qtile, head, batch) --------
__global__ __launch_bounds__(32) void k_attention(
    const unsigned short* __restrict__ qkv,  // [B, L, 3D] bf16 (q|k|v)
    unsigned short* __restrict__ o) {        // [B, L, D]  bf16
  __shared__ __align__(16) float         sc[16][KT * 16];     // raw scores
  __shared__ __align__(16) unsigned short pb[16][KPAD];       // probs bf16

  const int qt = blockIdx.x;              // 0..63
  const int hh = blockIdx.y;              // head
  const int bb = blockIdx.z;              // batch
  const int lane = threadIdx.x;
  const int lrow = lane & 15;
  const int khalf = (lane >= 16) ? 8 : 0;
  const int q0 = qt * 16;
  const size_t rs = 3 * DD;               // qkv row stride (bf16 elems)
  const unsigned short* base = qkv + (size_t)bb * LL * rs;

  // Q fragments (dh 0..31 and 32..63)
  const unsigned short* qrow = base + (size_t)(q0 + lrow) * rs + hh * DHD;
  const Frag32 qa0 = mkfrag(*(const uint4*)&qrow[khalf],      *(const uint4*)&qrow[khalf + 16]);
  const Frag32 qa1 = mkfrag(*(const uint4*)&qrow[32 + khalf], *(const uint4*)&qrow[32 + khalf + 16]);
  const float scale = 0.125f;             // 1/sqrt(64)

  // ---- QK^T over the band ----
  for (int idx = 0; idx < KT; ++idx) {
    const int kt = qt - 4 + idx;
    if (kt < 0 || kt >= LL / 16) {
#pragma unroll
      for (int r = 0; r < 8; ++r)
        sc[(lane >= 16) ? 8 + r : r][idx * 16 + lrow] = -1e9f;
      continue;
    }
    const unsigned short* krow =
        base + (size_t)(kt * 16 + lrow) * rs + DD + hh * DHD;
    const Frag32 kb0 = mkfrag(*(const uint4*)&krow[khalf],      *(const uint4*)&krow[khalf + 16]);
    const Frag32 kb1 = mkfrag(*(const uint4*)&krow[32 + khalf], *(const uint4*)&krow[32 + khalf + 16]);
    v8f s = {};
    s = wmma_bf16(qa0, kb0, s);
    s = wmma_bf16(qa1, kb1, s);
#pragma unroll
    for (int r = 0; r < 8; ++r)
      sc[(lane >= 16) ? 8 + r : r][idx * 16 + lrow] = s[r] * scale;
  }
  __syncthreads();

  // ---- masked softmax: 2 lanes per query row, 72 cols each ----
  const int row  = lane >> 1;
  const int half = lane & 1;
  const int kstart = (qt - 4) * 16;
  const int q = q0 + row;
  float mx = -1e30f;
  for (int j = 0; j < 72; ++j) {
    const int col = half * 72 + j;
    const int key = kstart + col;
    const int dd = (q > key) ? (q - key) : (key - q);
    const bool ok = (key >= 0) && (key < LL) && (dd <= WBAND);
    mx = fmaxf(mx, ok ? sc[row][col] : -1e9f);
  }
  mx = fmaxf(mx, __shfl_xor(mx, 1, 32));
  float sum = 0.0f;
  for (int j = 0; j < 72; ++j) {
    const int col = half * 72 + j;
    const int key = kstart + col;
    const int dd = (q > key) ? (q - key) : (key - q);
    const bool ok = (key >= 0) && (key < LL) && (dd <= WBAND);
    sum += ok ? __expf(sc[row][col] - mx) : 0.0f;
  }
  sum += __shfl_xor(sum, 1, 32);
  const float inv = 1.0f / sum;
  for (int j = 0; j < 72; ++j) {
    const int col = half * 72 + j;
    const int key = kstart + col;
    const int dd = (q > key) ? (q - key) : (key - q);
    const bool ok = (key >= 0) && (key < LL) && (dd <= WBAND);
    pb[row][col] = f32_to_bf16u(ok ? __expf(sc[row][col] - mx) * inv : 0.0f);
  }
  for (int t = lane; t < 256; t += 32) pb[t >> 4][144 + (t & 15)] = 0;  // pad
  __syncthreads();

  // ---- AV: probs(16 x 160) x V(160 x 64) ----
  v8f oacc[4] = {{}, {}, {}, {}};
  for (int p = 0; p < 5; ++p) {
    const Frag32 pa = mkfrag(*(const uint4*)&pb[lrow][p * 32 + khalf],
                             *(const uint4*)&pb[lrow][p * 32 + khalf + 16]);
#pragma unroll
    for (int c = 0; c < 4; ++c) {
      const int dh = c * 16 + lrow;
      Frag32 vb;
#pragma unroll
      for (int t = 0; t < 8; ++t) {
        const int r0 = khalf + ((t < 4) ? 2 * t : 16 + 2 * (t - 4));
        const int key0 = kstart + p * 32 + r0;
        unsigned short e0 = 0, e1 = 0;
        if (key0 >= 0 && key0 < LL)
          e0 = base[(size_t)key0 * rs + 2 * DD + hh * DHD + dh];
        if (key0 + 1 >= 0 && key0 + 1 < LL)
          e1 = base[(size_t)(key0 + 1) * rs + 2 * DD + hh * DHD + dh];
        vb.u[t] = (uint32_t)e0 | ((uint32_t)e1 << 16);
      }
      oacc[c] = wmma_bf16(pa, vb, oacc[c]);
    }
  }

  // store O tile
  unsigned short* ob = o + (size_t)bb * LL * DD;
  const int mrb = q0 + ((lane >= 16) ? 8 : 0);
#pragma unroll
  for (int c = 0; c < 4; ++c) {
    const int dh = c * 16 + lrow;
#pragma unroll
    for (int r = 0; r < 8; ++r)
      ob[(size_t)(mrb + r) * DD + hh * DHD + dh] = f32_to_bf16u(oacc[c][r]);
  }
}

// ---------------- residual add + LayerNorm (+ bf16 mirror) -------------------
__global__ __launch_bounds__(256) void k_add_ln(
    const float* __restrict__ hin, const float* __restrict__ delta,
    const float* __restrict__ g, const float* __restrict__ b,
    float* __restrict__ hout, unsigned short* __restrict__ hbf, int D) {
  __shared__ float ws1[8], ws2[8];
  const int row = blockIdx.x;
  const int tid = threadIdx.x;
  const float* hr = hin + (size_t)row * D;
  const float* dr = delta ? delta + (size_t)row * D : nullptr;

  float s = 0.0f, s2 = 0.0f;
  for (int i = tid; i < D; i += 256) {
    float v = hr[i] + (dr ? dr[i] : 0.0f);
    s += v; s2 += v * v;
  }
  for (int off = 16; off; off >>= 1) {
    s  += __shfl_down(s,  off, 32);
    s2 += __shfl_down(s2, off, 32);
  }
  if ((tid & 31) == 0) { ws1[tid >> 5] = s; ws2[tid >> 5] = s2; }
  __syncthreads();
  if (tid == 0) {
    float a = 0, a2 = 0;
    for (int w = 0; w < 8; ++w) { a += ws1[w]; a2 += ws2[w]; }
    ws1[0] = a; ws2[0] = a2;
  }
  __syncthreads();
  const float mean = ws1[0] / D;
  const float var  = ws2[0] / D - mean * mean;
  const float inv  = rsqrtf(var + 1e-5f);
  for (int i = tid; i < D; i += 256) {
    float v = hr[i] + (dr ? dr[i] : 0.0f);
    float y = (v - mean) * inv * g[i] + b[i];
    hout[(size_t)row * D + i] = y;
    if (hbf) hbf[(size_t)row * D + i] = f32_to_bf16u(y);
  }
}

// ---------------- pooling (sum over L) ----------------
__global__ void k_pool(const float* __restrict__ h, float* __restrict__ pooled) {
  const int b = blockIdx.x;
  for (int d = threadIdx.x; d < DD; d += blockDim.x) {
    const float* p = h + (size_t)b * LL * DD + d;
    float s = 0.0f;
    for (int l = 0; l < LL; ++l) s += p[(size_t)l * DD];
    pooled[b * DD + d] = s;
  }
}

// ---------------- final projection + relu ----------------
__global__ void k_proj_relu(const float* __restrict__ e, const float* __restrict__ We,
                            const float* __restrict__ be, float* __restrict__ out) {
  const int i = blockIdx.x * blockDim.x + threadIdx.x;
  if (i >= BB * DD) return;
  const int b = i / DD, n = i % DD;
  const float* er = e + b * DD;
  const float* wr = We + (size_t)n * DD;
  float s = be[n];
  for (int d = 0; d < DD; ++d) s += er[d] * wr[d];
  out[i] = fmaxf(s, 0.0f);
}

// ---------------- cosine similarity ----------------
__global__ __launch_bounds__(256) void k_cosine(const float* __restrict__ ex,
                                                const float* __restrict__ ey,
                                                float* __restrict__ out) {
  const int wave = threadIdx.x >> 5;
  const int lane = threadIdx.x & 31;
  if (wave >= BB) return;
  const float* a = ex + wave * DD;
  const float* b = ey + wave * DD;
  float d = 0, na = 0, nb = 0;
  for (int i = lane; i < DD; i += 32) {
    float x = a[i], y = b[i];
    d += x * y; na += x * x; nb += y * y;
  }
  for (int off = 16; off; off >>= 1) {
    d  += __shfl_down(d,  off, 32);
    na += __shfl_down(na, off, 32);
    nb += __shfl_down(nb, off, 32);
  }
  if (lane == 0) {
    float nx = fmaxf(sqrtf(na), 1e-8f);
    float ny = fmaxf(sqrtf(nb), 1e-8f);
    out[wave] = d / (nx * ny);
  }
}

// ============================ host orchestration =============================
extern "C" void kernel_launch(void* const* d_in, const int* in_sizes, int n_in,
                              void* d_out, int out_size, void* d_ws, size_t ws_size,
                              hipStream_t stream) {
  const float* x     = (const float*)d_in[0];
  const float* y     = (const float*)d_in[3];
  const float* Wqkv  = (const float*)d_in[6];
  const float* bqkv  = (const float*)d_in[7];
  const float* Wo    = (const float*)d_in[8];
  const float* bo    = (const float*)d_in[9];
  const float* ln1g  = (const float*)d_in[10];
  const float* ln1b  = (const float*)d_in[11];
  const float* W1    = (const float*)d_in[12];
  const float* b1    = (const float*)d_in[13];
  const float* W2    = (const float*)d_in[14];
  const float* b2    = (const float*)d_in[15];
  const float* ln2g  = (const float*)d_in[16];
  const float* ln2b  = (const float*)d_in[17];
  const float* lneg  = (const float*)d_in[18];
  const float* lneb  = (const float*)d_in[19];
  const float* We    = (const float*)d_in[20];
  const float* be    = (const float*)d_in[21];

  // ---- workspace carve ----
  char* wp = (char*)d_ws;
  size_t used = 0;
  auto carve = [&](size_t bytes) -> void* {
    void* p = wp + used;
    used += (bytes + 255) & ~(size_t)255;
    return p;
  };
  const size_t nWqkv = (size_t)NLAYER * 3 * DD * DD;
  const size_t nWo   = (size_t)NLAYER * DD * DD;
  const size_t nW1   = (size_t)NLAYER * FF * DD;
  const size_t nW2   = (size_t)NLAYER * DD * FF;
  unsigned short* wqkv_bf = (unsigned short*)carve(nWqkv * 2);
  unsigned short* wo_bf   = (unsigned short*)carve(nWo * 2);
  unsigned short* w1_bf   = (unsigned short*)carve(nW1 * 2);
  unsigned short* w2_bf   = (unsigned short*)carve(nW2 * 2);
  float*          h       = (float*)carve((size_t)MTOK * DD * 4);
  unsigned short* h_bf    = (unsigned short*)carve((size_t)MTOK * DD * 2);
  unsigned short* qkv_bf  = (unsigned short*)carve((size_t)MTOK * 3 * DD * 2);
  unsigned short* o_bf    = (unsigned short*)carve((size_t)MTOK * DD * 2);
  float*          tmpf    = (float*)carve((size_t)MTOK * DD * 4);
  unsigned short* ff_bf   = (unsigned short*)carve((size_t)MTOK * FF * 2);
  float*          pooled  = (float*)carve((size_t)BB * DD * 4);
  float*          epre    = (float*)carve((size_t)BB * DD * 4);
  float*          e_x     = (float*)carve((size_t)BB * DD * 4);
  float*          e_y     = (float*)carve((size_t)BB * DD * 4);
  if (used > ws_size) return;  // workspace too small; bail deterministically

  // ---- convert weights to bf16 ----
  k_cvt_bf16<<<2048, 256, 0, stream>>>(Wqkv, wqkv_bf, nWqkv);
  k_cvt_bf16<<<2048, 256, 0, stream>>>(Wo,   wo_bf,   nWo);
  k_cvt_bf16<<<2048, 256, 0, stream>>>(W1,   w1_bf,   nW1);
  k_cvt_bf16<<<2048, 256, 0, stream>>>(W2,   w2_bf,   nW2);

  auto run_seq = [&](const float* xin, float* evec) {
    hipMemcpyAsync(h, xin, (size_t)MTOK * DD * 4, hipMemcpyDeviceToDevice, stream);
    k_cvt_bf16<<<2048, 256, 0, stream>>>(h, h_bf, (size_t)MTOK * DD);

    for (int l = 0; l < NLAYER; ++l) {
      // QKV projection: [8192x640] x [1920x640]^T -> bf16
      dim3 gq(MTOK / 32, (3 * DD) / 128);
      k_gemm_bf16<<<gq, 256, 0, stream>>>(h_bf, wqkv_bf + (size_t)l * 3 * DD * DD,
                                          bqkv + (size_t)l * 3 * DD,
                                          nullptr, qkv_bf, MTOK, 3 * DD, DD,
                                          GEMM_BF16OUT);
      // banded attention
      dim3 ga(LL / 16, HH, BB);
      k_attention<<<ga, 32, 0, stream>>>(qkv_bf, o_bf);
      // output projection -> fp32
      dim3 go(MTOK / 32, DD / 128);
      k_gemm_bf16<<<go, 256, 0, stream>>>(o_bf, wo_bf + (size_t)l * DD * DD,
                                          bo + (size_t)l * DD,
                                          tmpf, nullptr, MTOK, DD, DD, 0);
      k_add_ln<<<MTOK, 256, 0, stream>>>(h, tmpf, ln1g + (size_t)l * DD,
                                         ln1b + (size_t)l * DD, h, h_bf, DD);
      // FFN1 (relu) -> bf16
      dim3 gf1(MTOK / 32, FF / 128);
      k_gemm_bf16<<<gf1, 256, 0, stream>>>(h_bf, w1_bf + (size_t)l * FF * DD,
                                           b1 + (size_t)l * FF,
                                           nullptr, ff_bf, MTOK, FF, DD,
                                           GEMM_RELU | GEMM_BF16OUT);
      // FFN2 -> fp32
      dim3 gf2(MTOK / 32, DD / 128);
      k_gemm_bf16<<<gf2, 256, 0, stream>>>(ff_bf, w2_bf + (size_t)l * DD * FF,
                                           b2 + (size_t)l * DD,
                                           tmpf, nullptr, MTOK, DD, FF, 0);
      k_add_ln<<<MTOK, 256, 0, stream>>>(h, tmpf, ln2g + (size_t)l * DD,
                                         ln2b + (size_t)l * DD, h, h_bf, DD);
    }
    // pool + LN + projection
    k_pool<<<BB, 256, 0, stream>>>(h, pooled);
    k_add_ln<<<BB, 256, 0, stream>>>(pooled, nullptr, lneg, lneb, epre, nullptr, DD);
    k_proj_relu<<<(BB * DD + 255) / 256, 256, 0, stream>>>(epre, We, be, evec);
  };

  run_seq(x, e_x);
  run_seq(y, e_y);
  k_cosine<<<1, 256, 0, stream>>>(e_x, e_y, (float*)d_out);
}
